// CDEModel_73658689127106
// MI455X (gfx1250) — matrix operand order repaired and checked by reading, not compiled, not run on hardware
//
#include <hip/hip_runtime.h>

typedef __attribute__((ext_vector_type(16))) _Float16 v16h;
typedef __attribute__((ext_vector_type(8)))  float    v8f;
typedef __attribute__((ext_vector_type(4)))  float    v4f;

#define NTHREADS 256
#define BTILE    32
#define NWG      128      // 4096 / BTILE

#define LATENT   64
#define HIDDEN   128
#define XDIM     9
#define LKNOTS   128
#define NINT     127
#define SDIM     32

// ---------------- LDS layout (bytes) ----------------
#define W2_STRIDE 136     // 128 K + 8 pad halves -> 272B rows, conflict-free b128
#define W1_STRIDE 72      // 64 K + 8 pad halves
#define ZA_STRIDE 72
#define H_STRIDE  136
#define DX_STRIDE 10

#define OFF_W2   0
#define SZ_W2    (576*W2_STRIDE*2)
#define OFF_W1   (OFF_W2 + SZ_W2)
#define SZ_W1    (HIDDEN*W1_STRIDE*2)
#define OFF_DEC  (OFF_W1 + SZ_W1)
#define SZ_DEC   (SDIM*W1_STRIDE*2)
#define OFF_M    (OFF_DEC + SZ_DEC)
#define SZ_M     (BTILE*LKNOTS*XDIM*2)
#define OFF_ZA   (OFF_M + SZ_M)
#define SZ_ZA    (BTILE*ZA_STRIDE*2)
#define OFF_H    (OFF_ZA + SZ_ZA)
#define SZ_H     (BTILE*H_STRIDE*2)
#define OFF_DX   (OFF_H + SZ_H)
#define SZ_DX    (5*BTILE*DX_STRIDE*4)
#define OFF_B1   (OFF_DX + SZ_DX)
#define SZ_B1    (HIDDEN*4)
#define OFF_B2   (OFF_B1 + SZ_B1)
#define SZ_B2    (576*4)
#define OFF_DB   (OFF_B2 + SZ_B2)
#define SZ_DB    (SDIM*4)
#define OFF_CP   (OFF_DB + SZ_DB)
#define SZ_CP    (128*4)
#define SMEM_BYTES (OFF_CP + SZ_CP)   // ~276.7 KB < 320 KB

#define ZERO8 v8f{0.f,0.f,0.f,0.f,0.f,0.f,0.f,0.f}

// ---------------- WMMA fragment helpers ----------------
// 16-bit A 16x32: lane<16 -> K [0..15]; lane>=16 -> K [16..31]
__device__ __forceinline__ v16h load_a16(const _Float16* row, int klo) {
  union { v16h v; v4f q[2]; } u;
  u.q[0] = *(const v4f*)(row + klo);
  u.q[1] = *(const v4f*)(row + klo + 8);
  return u.v;
}
// 16-bit B 32x16 (stored K-contiguous per column): lane<16 -> K [0..15]; lane>=16 -> K [16..31]
__device__ __forceinline__ v16h load_b16(const _Float16* row, int k0) {
  union { v16h v; v4f q[2]; } u;
  u.q[0] = *(const v4f*)(row + k0);
  u.q[1] = *(const v4f*)(row + k0 + 8);
  return u.v;
}
__device__ __forceinline__ v8f wmma_f16(v16h a, v16h b, v8f c) {
  return __builtin_amdgcn_wmma_f32_16x16x32_f16(false, a, false, b, (short)0, c, false, false);
}

__device__ __forceinline__ float getx(const float* a, const float* t, size_t gb, int l, int x) {
  return (x == 0) ? t[gb*LKNOTS + l] : a[(gb*LKNOTS + l)*8 + (x - 1)];
}

// decoder: out[b, iv, :] = zA @ decW + decb  via 4 WMMA tiles (waves 0..3)
__device__ __forceinline__ void decode_knot(const _Float16* sZA, const _Float16* sDec,
                                            const float* sDB, float* __restrict__ out,
                                            int b0, int iv, int wv, int c, int hlf) {
  int MtD = wv >> 1, NtD = wv & 1;
  const _Float16* arow = sZA + (MtD*16 + c)*ZA_STRIDE;
  const _Float16* brow = sDec + (NtD*16 + c)*W1_STRIDE;
  v8f accD = ZERO8;
#pragma unroll
  for (int kk = 0; kk < 2; kk++) {
    v16h af = load_a16(arow, kk*32 + (hlf ? 16 : 0));
    v16h bf = load_b16(brow, kk*32 + hlf*16);
    accD = wmma_f16(af, bf, accD);
  }
  float dbias = sDB[NtD*16 + c];
#pragma unroll
  for (int r = 0; r < 8; r++) {
    size_t gb = (size_t)(b0 + MtD*16 + 8*hlf + r);
    out[(gb*LKNOTS + (size_t)iv)*SDIM + NtD*16 + c] = accD[r] + dbias;
  }
}

__global__ __launch_bounds__(NTHREADS, 1)
void ncde_wmma_kernel(const float* __restrict__ s,   const float* __restrict__ a,
                      const float* __restrict__ t,   const float* __restrict__ encW,
                      const float* __restrict__ encb,const float* __restrict__ fW1,
                      const float* __restrict__ fb1, const float* __restrict__ fW2,
                      const float* __restrict__ fb2, const float* __restrict__ decW,
                      const float* __restrict__ decb,float* __restrict__ out)
{
  extern __shared__ char smem[];
  _Float16* sW2  = (_Float16*)(smem + OFF_W2);   // [576][136]  n = x*64 + latent, K-major rows
  _Float16* sW1  = (_Float16*)(smem + OFF_W1);   // [128][72]
  _Float16* sDec = (_Float16*)(smem + OFF_DEC);  // [32][72]
  _Float16* sM   = (_Float16*)(smem + OFF_M);    // [32][128][9] spline 2nd derivatives
  _Float16* sZA  = (_Float16*)(smem + OFF_ZA);   // [32][72]  z in f16 (GEMM1 A)
  _Float16* sH   = (_Float16*)(smem + OFF_H);    // [32][136] hidden (GEMM2 A)
  float* sDX  = (float*)(smem + OFF_DX);         // [5][32][10] dX at u = 0,.25,.5,.75,1
  float* sB1  = (float*)(smem + OFF_B1);
  float* sB2r = (float*)(smem + OFF_B2);
  float* sDB  = (float*)(smem + OFF_DB);
  float* sCP  = (float*)(smem + OFF_CP);         // Thomas c' table (shared by all systems)

  const int tid  = threadIdx.x;
  const int lane = tid & 31;
  const int wv   = tid >> 5;
  const int c    = lane & 15;
  const int hlf  = lane >> 4;
  const int b0   = blockIdx.x * BTILE;

  // ---- phase 0: stage weights (f32 -> f16, reordered) into LDS ----
  for (int idx = tid; idx < HIDDEN*LATENT; idx += NTHREADS) {
    int n = idx >> 6, k = idx & 63;
    sW1[n*W1_STRIDE + k] = (_Float16)fW1[k*HIDDEN + n];
  }
  for (int idx = tid; idx < 576*HIDDEN; idx += NTHREADS) {
    int n = idx >> 7, k = idx & 127;
    int x = n >> 6, l = n & 63;                       // x-major: n = x*64 + l
    sW2[n*W2_STRIDE + k] = (_Float16)fW2[k*576 + l*XDIM + x];
  }
  for (int idx = tid; idx < SDIM*LATENT; idx += NTHREADS) {
    int n = idx >> 6, k = idx & 63;
    sDec[n*W1_STRIDE + k] = (_Float16)decW[k*SDIM + n];
  }
  for (int n = tid; n < HIDDEN; n += NTHREADS) sB1[n] = fb1[n];
  for (int n = tid; n < 576; n += NTHREADS) { int x = n >> 6, l = n & 63; sB2r[n] = fb2[l*XDIM + x]; }
  if (tid < SDIM) sDB[tid] = decb[tid];
  if (tid == 0) {                                     // constant-tridiag Thomas c' table
    float cprev = 0.f;
    for (int i = 0; i < 126; i++) { float cv = 1.f / (4.f - cprev); sCP[i] = cv; cprev = cv; }
  }
  __syncthreads();

  // ---- phase 1: encoder z0 -> f32 scratch (aliases sH, consumed before GEMM1) ----
  float* z0tmp = (float*)sH;
  for (int idx = tid; idx < BTILE*LATENT; idx += NTHREADS) {
    int bb = idx >> 6, j = idx & 63;
    size_t gb = (size_t)(b0 + bb);
    const float* sr = s + gb*(size_t)LKNOTS*SDIM;
    const float* ar = a + gb*(size_t)LKNOTS*8;
    float acc = encb[j];
    for (int k = 0; k < SDIM; k++) acc = fmaf(sr[k], encW[k*LATENT + j], acc);
    for (int k = 0; k < 8; k++)    acc = fmaf(ar[k], encW[(SDIM + k)*LATENT + j], acc);
    acc = fmaf(t[gb*LKNOTS], encW[40*LATENT + j], acc);
    z0tmp[idx] = acc;
  }

  // ---- phase 2: natural cubic spline: Thomas solve per (b, channel) -> sM (f16) ----
  for (int sys = tid; sys < BTILE*XDIM; sys += NTHREADS) {
    int bb = sys / XDIM, x = sys % XDIM;
    size_t gb = (size_t)(b0 + bb);
    float dp[126];
    float xm2 = getx(a, t, gb, 0, x);
    float xm1 = getx(a, t, gb, 1, x);
    float dprev = 0.f;
    for (int i = 0; i < 126; i++) {
      float xc  = getx(a, t, gb, i + 2, x);
      float rhs = 6.f * (xc - 2.f*xm1 + xm2);
      dprev = (rhs - dprev) * sCP[i];
      dp[i] = dprev;
      xm2 = xm1; xm1 = xc;
    }
    sM[(bb*LKNOTS + 0)*XDIM + x]   = (_Float16)0.f;   // natural BCs
    sM[(bb*LKNOTS + 127)*XDIM + x] = (_Float16)0.f;
    float mnext = 0.f;
    for (int i = 125; i >= 0; i--) {
      float mi = dp[i] - sCP[i]*mnext;
      sM[(bb*LKNOTS + i + 1)*XDIM + x] = (_Float16)mi;
      mnext = mi;
    }
  }
  __syncthreads();

  // ---- per-lane persistent RK4 state (ownership matches GEMM2 D-tile layout) ----
  const int Mtile   = wv >> 2;          // batch tile 0..1
  const int jl      = wv & 3;           // latent 16-chunk 0..3
  const int myl     = jl*16 + c;        // latent index 0..63
  const int bb_base = Mtile*16 + 8*hlf; // + r = local batch row
  float zreg[8], ksum[8], b2v[XDIM];
#pragma unroll
  for (int r = 0; r < 8; r++) zreg[r] = z0tmp[(bb_base + r)*LATENT + myl];
#pragma unroll
  for (int x = 0; x < XDIM; x++) b2v[x] = sB2r[x*64 + myl];
#pragma unroll
  for (int r = 0; r < 8; r++) {
    ksum[r] = 0.f;
    sZA[(bb_base + r)*ZA_STRIDE + myl] = (_Float16)zreg[r];
  }

  // ---- hoisted loop-invariant per-lane addresses / scalars ----
  const int alo  = hlf ? 16 : 0;         // A-fragment K-chunk base
  const int bko  = hlf * 16;             // B-fragment K base
  const int col0 = (wv & 3)*16 + c;      // GEMM1 output columns for this wave
  const int col1 = col0 + 64;
  const _Float16* zarow  = sZA + (Mtile*16 + c)*ZA_STRIDE;  // GEMM1 A row
  const _Float16* hrow   = sH  + (Mtile*16 + c)*H_STRIDE;   // GEMM2 A row
  const _Float16* w1row0 = sW1 + col0*W1_STRIDE;
  const _Float16* w1row1 = sW1 + col1*W1_STRIDE;
  const float bias0 = sB1[col0];
  const float bias1 = sB1[col1];
  const _Float16* w2row[XDIM];
#pragma unroll
  for (int x = 0; x < XDIM; x++) w2row[x] = sW2 + ((4*x + jl)*16 + c)*W2_STRIDE;
  _Float16* hst  = sH  + bb_base*H_STRIDE;            // + r*H_STRIDE + col
  _Float16* zast = sZA + bb_base*ZA_STRIDE + myl;     // + r*ZA_STRIDE
  const float* dxbase = sDX + bb_base*DX_STRIDE;      // + uidx*BTILE*DX_STRIDE + r*10 + x

  // ---- main scan over intervals ----
#pragma unroll 1
  for (int iv = 0; iv < NINT; iv++) {
    __syncthreads();
    if (wv < 4) decode_knot(sZA, sDec, sDB, out, b0, iv, wv, c, hlf);

    // dX(u) for the 5 distinct u samples of this interval
    for (int p = tid; p < BTILE*XDIM; p += NTHREADS) {
      int bb = p / XDIM, x = p % XDIM;
      size_t gb = (size_t)(b0 + bb);
      float x0 = getx(a, t, gb, iv, x);
      float x1 = getx(a, t, gb, iv + 1, x);
      float m0 = (float)sM[(bb*LKNOTS + iv)*XDIM + x];
      float m1 = (float)sM[(bb*LKNOTS + iv + 1)*XDIM + x];
      float bc  = (x1 - x0) - (2.f*m0 + m1) * (1.f/6.f);
      float cc2 = m0;                 // 2c
      float dc3 = 0.5f*(m1 - m0);     // 3d
#pragma unroll
      for (int uu = 0; uu < 5; uu++) {
        float u = 0.25f * (float)uu;
        sDX[(uu*BTILE + bb)*DX_STRIDE + x] = bc + (cc2 + dc3*u)*u;
      }
    }
    // warm next interval's action rows (gfx1250 global_prefetch)
    if (iv + 2 <= NINT) {
      size_t gb = (size_t)(b0 + (tid & 31));
      __builtin_prefetch(&a[(gb*LKNOTS + iv + 2)*8], 0, 1);
    }

#pragma unroll 1
    for (int sub = 0; sub < 2; sub++) {
#pragma unroll
      for (int e = 0; e < 4; e++) {
        __syncthreads();
        // ---- GEMM1: H = relu(z @ W1 + b1), M=32 N=128 K=64 (2 interleaved chains) ----
        {
          v16h af0 = load_a16(zarow, alo);
          v16h af1 = load_a16(zarow, 32 + alo);
          v8f acc0 = ZERO8;
          v8f acc1 = ZERO8;
          acc0 = wmma_f16(af0, load_b16(w1row0, bko),      acc0);
          acc1 = wmma_f16(af0, load_b16(w1row1, bko),      acc1);
          acc0 = wmma_f16(af1, load_b16(w1row0, 32 + bko), acc0);
          acc1 = wmma_f16(af1, load_b16(w1row1, 32 + bko), acc1);
#pragma unroll
          for (int r = 0; r < 8; r++) {
            float v0 = acc0[r] + bias0; v0 = v0 > 0.f ? v0 : 0.f;
            float v1 = acc1[r] + bias1; v1 = v1 > 0.f ? v1 : 0.f;
            hst[r*H_STRIDE + col0] = (_Float16)v0;
            hst[r*H_STRIDE + col1] = (_Float16)v1;
          }
        }
        __syncthreads();
        // ---- GEMM2 (M=32 N=576 K=128, x-major N): K-outer over 9 independent
        //      accumulator chains -> no WMMA->WMMA hazard stalls ----
        {
          const int uidx = sub*2 + ((e == 0) ? 0 : ((e == 3) ? 2 : 1));
          const float* dxp = dxbase + uidx*(BTILE*DX_STRIDE);
          v8f acc[XDIM];
#pragma unroll
          for (int x = 0; x < XDIM; x++) acc[x] = ZERO8;
#pragma unroll
          for (int kk = 0; kk < 4; kk++) {
            v16h af = load_a16(hrow, kk*32 + alo);
#pragma unroll
            for (int x = 0; x < XDIM; x++)
              acc[x] = wmma_f16(af, load_b16(w2row[x], kk*32 + bko), acc[x]);
          }
          // contraction with dX + bias, then RK4 bookkeeping in registers
          float dz[8];
#pragma unroll
          for (int r = 0; r < 8; r++) {
            float tacc = 0.f;
#pragma unroll
            for (int x = 0; x < XDIM; x++)
              tacc = fmaf(acc[x][r] + b2v[x], dxp[r*DX_STRIDE + x], tacc);
            dz[r] = tacc;
          }
          const float kw = (e == 0 || e == 3) ? 1.f : 2.f;
          const float al = (e <= 1) ? 0.25f : 0.5f;  // h=0.5: z+0.25k1, z+0.25k2, z+0.5k3
#pragma unroll
          for (int r = 0; r < 8; r++) {
            float d = dz[r];
            if (e == 0) ksum[r] = d; else ksum[r] = fmaf(kw, d, ksum[r]);
            float ze;
            if (e == 3) { zreg[r] = fmaf(1.f/12.f, ksum[r], zreg[r]); ze = zreg[r]; }
            else          ze = fmaf(al, d, zreg[r]);
            zast[r*ZA_STRIDE] = (_Float16)ze;
          }
        }
      }
    }
  }

  __syncthreads();
  if (wv < 4) decode_knot(sZA, sDec, sDB, out, b0, NINT, wv, c, hlf);
}

extern "C" void kernel_launch(void* const* d_in, const int* in_sizes, int n_in,
                              void* d_out, int out_size, void* d_ws, size_t ws_size,
                              hipStream_t stream) {
  (void)in_sizes; (void)n_in; (void)d_ws; (void)ws_size; (void)out_size;
  const float* s    = (const float*)d_in[0];
  const float* a    = (const float*)d_in[1];
  const float* t    = (const float*)d_in[2];
  const float* encW = (const float*)d_in[3];
  const float* encb = (const float*)d_in[4];
  const float* fW1  = (const float*)d_in[5];
  const float* fb1  = (const float*)d_in[6];
  const float* fW2  = (const float*)d_in[7];
  const float* fb2  = (const float*)d_in[8];
  const float* decW = (const float*)d_in[9];
  const float* decb = (const float*)d_in[10];
  float* out = (float*)d_out;

  (void)hipFuncSetAttribute(reinterpret_cast<const void*>(ncde_wmma_kernel),
                            hipFuncAttributeMaxDynamicSharedMemorySize,
                            (int)SMEM_BYTES);
  ncde_wmma_kernel<<<dim3(NWG), dim3(NTHREADS), SMEM_BYTES, stream>>>(
      s, a, t, encW, encb, fW1, fb1, fW2, fb2, decW, decb, out);
}
